// toy_attn_75866302317145
// MI455X (gfx1250) — compile-verified, matrix operand order checked by reading, not run on hardware
//
#include <hip/hip_runtime.h>
#include <hip/hip_bf16.h>

typedef __attribute__((ext_vector_type(16))) __bf16 v16bf;
typedef __attribute__((ext_vector_type(8)))  float  v8f;

#define S_LEN 2048
#define DM    1024
#define NH    16
#define DH    64
#define NB    2
#define NROW  (NB * S_LEN)   // 4096 token rows

// ---- WMMA helpers (wave32, CDNA5 layouts, ISA 7.12.2) ------------------------
// A-operand 16x32 bf16: lane L holds row M=L&15; half-index e (0..15) maps to
//   K = (L&16 ? 8 : 0) + e            for e in [0,8)   (VGPR0-3)
//   K = 16 + (L&16 ? 8 : 0) + (e-8)   for e in [8,16)  (VGPR4-7)
// B-operand 32x16 mirrors A with N as the lane row (column-major gather).

__device__ __forceinline__ v16bf load_op16(const __bf16* p, int kb8) {
  union { v16bf v; uint4 u[2]; } r;
  r.u[0] = *(const uint4*)(p + kb8);
  r.u[1] = *(const uint4*)(p + 16 + kb8);
  return r.v;
}

__device__ __forceinline__ v16bf cvt_f32_op16(const float* p, int kb8) {
  v16bf a;
#pragma unroll
  for (int j = 0; j < 8; ++j) a[j] = (__bf16)p[kb8 + j];
#pragma unroll
  for (int j = 0; j < 8; ++j) a[j + 8] = (__bf16)p[16 + kb8 + j];
  return a;
}

__device__ __forceinline__ v8f wmma_bf16(v16bf a, v16bf b, v8f c) {
  return __builtin_amdgcn_wmma_f32_16x16x32_bf16(false, a, false, b, (short)0, c,
                                                 false, false);
}

// ---- Kernel 1: one-time fp32->bf16 conversions -------------------------------
// wtq/wtk/wtv: [n = h*64+d][m] bf16 (1024x1024).  wot: [h][nm][d] bf16.
// residb: [4096][1024] bf16 copy of resid (A-operand source for QKV).
__global__ void prep_inputs(const float* __restrict__ WQ, const float* __restrict__ WK,
                            const float* __restrict__ WV, const float* __restrict__ WO,
                            const float* __restrict__ resid,
                            __bf16* wtq, __bf16* wtk, __bf16* wtv, __bf16* wot,
                            __bf16* residb) {
  int idx   = blockIdx.x * blockDim.x + threadIdx.x;   // 0 .. 8M-1
  int which = idx >> 20;
  int t     = idx & 0xFFFFF;
  if (which < 3) {
    const float* W  = (which == 0) ? WQ : (which == 1) ? WK : WV;
    __bf16*      wt = (which == 0) ? wtq : (which == 1) ? wtk : wtv;
    int n = t >> 10, m = t & 1023;                     // n = h*64+d
    wt[t] = (__bf16)W[(size_t)(n >> 6) * (DM * DH) + (size_t)m * DH + (n & 63)];
  } else if (which == 3) {
    int h = t >> 16, nm = (t >> 6) & 1023, d = t & 63;
    wot[t] = (__bf16)WO[(size_t)h * (DH * DM) + (size_t)d * DM + nm];
  } else {
    size_t i = (size_t)idx - (4u << 20);               // 0 .. 4M-1
    residb[i] = (__bf16)resid[i];
  }
}

// ---- Kernel 2: QKV projection (bf16 WMMA, fp32 accum, +bias) -----------------
// grid.x = 256 m-tiles of 16 rows; grid.y = 48 (mat=0..2, 16 n-groups of 64).
__global__ void __launch_bounds__(128)
qkv_proj(const __bf16* __restrict__ residb,
         const __bf16* __restrict__ wtq, const __bf16* __restrict__ wtk,
         const __bf16* __restrict__ wtv,
         const float* __restrict__ bQ, const float* __restrict__ bK,
         const float* __restrict__ bV,
         __bf16* qb, __bf16* kb, __bf16* vt) {
  const int lane = threadIdx.x & 31;
  const int wave = threadIdx.x >> 5;
  const int kb8  = (lane & 16) ? 8 : 0;
  const int hi8  = (lane & 16) ? 8 : 0;
  const int mrow = lane & 15;
  const int mat  = blockIdx.y >> 4;                       // 0=Q 1=K 2=V
  const int n0   = ((blockIdx.y & 15) * 4 + wave) * 16;
  const int col  = n0 + (lane & 15);                      // h*64+d

  const __bf16* wt   = (mat == 0) ? wtq : (mat == 1) ? wtk : wtv;
  const float*  bias = (mat == 0) ? bQ  : (mat == 1) ? bK  : bV;
  __bf16*       outp = (mat == 0) ? qb  : kb;

  const __bf16* arow = residb + (size_t)(blockIdx.x * 16 + mrow) * DM;
  const __bf16* brow = wt + (size_t)col * DM;

  v8f acc = {};
  for (int kk = 0; kk < DM / 32; ++kk) {
    v16bf a  = load_op16(arow + kk * 32, kb8);
    v16bf bm = load_op16(brow + kk * 32, kb8);
    acc = wmma_bf16(a, bm, acc);
  }
  const float bval = bias[col];
#pragma unroll
  for (int r = 0; r < 8; ++r) {
    int   orow = blockIdx.x * 16 + r + hi8;               // global token row
    float val  = acc[r] + bval;
    if (mat < 2) {
      outp[(size_t)orow * DM + col] = (__bf16)val;
    } else {                                              // V: store transposed [b][h][d][s]
      int b = orow >> 11, s = orow & (S_LEN - 1);
      vt[((size_t)(b * NH + (col >> 6)) * DH + (col & 63)) * S_LEN + s] = (__bf16)val;
    }
  }
}

// ---- Kernel 3: attention core ------------------------------------------------
// One WG (4 waves) per (b*H+h, 16-query tile). 128KB LDS score stripe
// (CDNA5: 320KB LDS/WGP -> 2 workgroups resident).
__global__ void __launch_bounds__(128)
attn_core(const __bf16* __restrict__ qb, const __bf16* __restrict__ kb,
          const __bf16* __restrict__ vt, __bf16* zb, float* pattern) {
  extern __shared__ float smem[];
  float* sc  = smem;               // [16][S_LEN]
  float* red = smem + 16 * S_LEN;  // [16][8]

  const int bh = blockIdx.x;       // b*16+h
  const int qt = blockIdx.y;       // query tile 0..127
  const int b  = bh >> 4;
  const int h  = bh & 15;
  const int q0 = qt * 16;
  const int kmax = q0 + 16;        // causal key extent for this tile
  const int lane = threadIdx.x & 31;
  const int wave = threadIdx.x >> 5;
  const int kb8  = (lane & 16) ? 8 : 0;
  const int hi8  = (lane & 16) ? 8 : 0;
  const int mrow = lane & 15;

  // Q tile A-operand (d=0..63 -> two K=32 chunks), kept in registers
  const __bf16* qrow = qb + (size_t)(b * S_LEN + q0 + mrow) * DM + h * DH;
  v16bf aq0 = load_op16(qrow, kb8);
  v16bf aq1 = load_op16(qrow + 32, kb8);

  // Phase 1: scores = (Q K^T) * sqrt(d)  (reference divides by d^-1/2), causal mask
  for (int kt = wave; kt <= qt; kt += 4) {
    const __bf16* krow = kb + (size_t)(b * S_LEN + kt * 16 + mrow) * DM + h * DH;
    if (kt + 4 <= qt) {            // software prefetch next K tile (global_prefetch_b8)
      __builtin_prefetch(krow + (size_t)64 * DM, 0, 1);
    }
    v16bf bk0 = load_op16(krow, kb8);
    v16bf bk1 = load_op16(krow + 32, kb8);
    v8f c = {};
    c = wmma_bf16(aq0, bk0, c);
    c = wmma_bf16(aq1, bk1, c);
    const int ncol = kt * 16 + (lane & 15);
#pragma unroll
    for (int r = 0; r < 8; ++r) {
      int   mq  = r + hi8;
      float vsc = c[r] * 8.0f;                     // * sqrt(64)
      if (ncol > q0 + mq) vsc = -__builtin_inff(); // causal
      sc[mq * S_LEN + ncol] = vsc;
    }
  }
  __syncthreads();

  // Phase 2: softmax per row (8 threads per row), write pattern (zero-pad rest)
  const int row = threadIdx.x >> 3;
  const int sub = threadIdx.x & 7;
  float m = -__builtin_inff();
  for (int c0 = sub; c0 < kmax; c0 += 8) m = fmaxf(m, sc[row * S_LEN + c0]);
  red[row * 8 + sub] = m;
  __syncthreads();
  m = -__builtin_inff();
#pragma unroll
  for (int j = 0; j < 8; ++j) m = fmaxf(m, red[row * 8 + j]);
  __syncthreads();
  float l = 0.0f;
  for (int c0 = sub; c0 < kmax; c0 += 8) {
    float e = __expf(sc[row * S_LEN + c0] - m);
    sc[row * S_LEN + c0] = e;
    l += e;
  }
  red[row * 8 + sub] = l;
  __syncthreads();
  l = 0.0f;
#pragma unroll
  for (int j = 0; j < 8; ++j) l += red[row * 8 + j];
  const float inv  = 1.0f / l;
  const int   kpad = (kmax + 31) & ~31;            // K=32-padded extent for PV
  float* prow = pattern + ((size_t)bh * S_LEN + q0 + row) * S_LEN;
  for (int c0 = sub; c0 < S_LEN; c0 += 8) {
    float v;
    if (c0 < kmax) { v = sc[row * S_LEN + c0] * inv; sc[row * S_LEN + c0] = v; }
    else           { v = 0.0f; if (c0 < kpad) sc[row * S_LEN + c0] = 0.0f; }
    prow[c0] = v;                                  // full row incl. zero upper triangle
  }
  __syncthreads();

  // Phase 3: z = P @ V ; each wave owns one 16-wide d-tile
  const int d0 = wave * 16;
  const __bf16* vrow = vt + ((size_t)bh * DH + d0 + (lane & 15)) * S_LEN;
  v8f acc = {};
  const int nk32 = kpad >> 5;
  for (int kk = 0; kk < nk32; ++kk) {
    if (kk + 1 < nk32) {
      __builtin_prefetch(vrow + (kk + 1) * 32, 0, 1);
    }
    v16bf ap = cvt_f32_op16(sc + mrow * S_LEN + kk * 32, kb8);
    v16bf bv = load_op16(vrow + kk * 32, kb8);
    acc = wmma_bf16(ap, bv, acc);
  }
  const int col = h * DH + d0 + (lane & 15);
#pragma unroll
  for (int r = 0; r < 8; ++r) {
    int orow = b * S_LEN + q0 + r + hi8;
    zb[(size_t)orow * DM + col] = (__bf16)acc[r];
  }
}

// ---- Kernel 4: out_heads = z @ W_O per head; out = sum_h out_heads -----------
// grid.x = 256 m-tiles; grid.y = 16 n-groups of 64; 4 waves x 16-col tiles.
__global__ void __launch_bounds__(128)
out_proj(const __bf16* __restrict__ zb, const __bf16* __restrict__ wot,
         float* out, float* out_heads) {
  const int lane  = threadIdx.x & 31;
  const int wave  = threadIdx.x >> 5;
  const int kb8   = (lane & 16) ? 8 : 0;
  const int hi8   = (lane & 16) ? 8 : 0;
  const int mrow  = lane & 15;
  const int n0    = blockIdx.y * 64 + wave * 16;
  const int nm    = n0 + (lane & 15);
  const int grow0 = blockIdx.x * 16;
  const __bf16* arow = zb + (size_t)(grow0 + mrow) * DM;

  v8f csum = {};
  for (int h = 0; h < NH; ++h) {
    const __bf16* ar = arow + h * DH;
    const __bf16* br = wot + ((size_t)h * DM + nm) * DH;
    v16bf a0 = load_op16(ar, kb8);
    v16bf a1 = load_op16(ar + 32, kb8);
    v16bf b0 = load_op16(br, kb8);
    v16bf b1 = load_op16(br + 32, kb8);
    v8f c = {};
    c = wmma_bf16(a0, b0, c);
    c = wmma_bf16(a1, b1, c);
    csum += c;
#pragma unroll
    for (int r = 0; r < 8; ++r) {
      int grow = grow0 + r + hi8;
      out_heads[((size_t)grow * NH + h) * DM + nm] = c[r];
    }
  }
#pragma unroll
  for (int r = 0; r < 8; ++r) {
    int grow = grow0 + r + hi8;
    out[(size_t)grow * DM + nm] = csum[r];
  }
}

// ---- Host launch -------------------------------------------------------------
extern "C" void kernel_launch(void* const* d_in, const int* in_sizes, int n_in,
                              void* d_out, int out_size, void* d_ws, size_t ws_size,
                              hipStream_t stream) {
  (void)in_sizes; (void)n_in; (void)out_size; (void)ws_size;
  const float* resid = (const float*)d_in[0];
  const float* WQ    = (const float*)d_in[1];
  const float* WK    = (const float*)d_in[2];
  const float* WV    = (const float*)d_in[3];
  const float* WO    = (const float*)d_in[4];
  const float* bQ    = (const float*)d_in[5];
  const float* bK    = (const float*)d_in[6];
  const float* bV    = (const float*)d_in[7];
  // d_in[8] = b_O is never added by the reference.

  char* ws = (char*)d_ws;                          // 48 MB used, all fully written
  __bf16* wtq    = (__bf16*)(ws);                  // 2 MB  [1024][1024]
  __bf16* wtk    = (__bf16*)(ws + (2u  << 20));    // 2 MB
  __bf16* wtv    = (__bf16*)(ws + (4u  << 20));    // 2 MB
  __bf16* wot    = (__bf16*)(ws + (6u  << 20));    // 2 MB  [16][1024][64]
  __bf16* qb     = (__bf16*)(ws + (8u  << 20));    // 8 MB  [4096][1024]
  __bf16* kb     = (__bf16*)(ws + (16u << 20));    // 8 MB  [4096][1024]
  __bf16* vt     = (__bf16*)(ws + (24u << 20));    // 8 MB  [2][16][64][2048]
  __bf16* zb     = (__bf16*)(ws + (32u << 20));    // 8 MB  [4096][1024]
  __bf16* residb = (__bf16*)(ws + (40u << 20));    // 8 MB  [4096][1024]

  float* out       = (float*)d_out;
  float* out_heads = out + (size_t)NROW * DM;                 // [4096][16][1024]
  float* pattern   = out_heads + (size_t)NROW * NH * DM;      // [32][2048][2048]

  prep_inputs<<<32768, 256, 0, stream>>>(WQ, WK, WV, WO, resid,
                                         wtq, wtk, wtv, wot, residb);
  qkv_proj<<<dim3(256, 48), 128, 0, stream>>>(residb, wtq, wtk, wtv, bQ, bK, bV,
                                              qb, kb, vt);
  const size_t lds_bytes = (size_t)(16 * S_LEN + 16 * 8) * sizeof(float); // 128.5 KB
  attn_core<<<dim3(32, 128), 128, lds_bytes, stream>>>(qb, kb, vt, zb, pattern);
  out_proj<<<dim3(256, 16), 128, 0, stream>>>(zb, wot, out, out_heads);
}
